// MemoryAsGateLayer_79293686219152
// MI455X (gfx1250) — matrix operand order repaired and checked by reading, not compiled, not run on hardware
//
#include <hip/hip_runtime.h>
#include <hip/hip_bf16.h>

// ---------------------------------------------------------------------------
// MemoryAsGate transformer block for MI455X (gfx1250, wave32, WMMA).
// All GEMMs via v_wmma_f32_16x16x32_f16 (f16 inputs, f32 accum).
// All WMMA operands are contiguous-per-lane (global_load_b128 pairs):
//   - weights transpose-cast to N x K f16 once per call (LDS-tiled)
//   - V transposed to dim-major once per call
// GEMM register-blocks 16x64 per wave and software-pipelines the K loop
// (double-buffered fragments) so global_load_b128 overlaps WMMA issue.
// Attention: flash-style online softmax; P tile goes C-layout -> LDS ->
// A-layout (ds_store/ds_load + s_wait_dscnt).
// ---------------------------------------------------------------------------

typedef __attribute__((ext_vector_type(16))) _Float16 v16h;
typedef __attribute__((ext_vector_type(8)))  float    v8f;

#define DIM      512
#define SEQ      4096
#define HEADS    8
#define HEAD_DIM 64
#define WINDOW   256
#define MEM_H    256
#define FFN_H    2048

__device__ __forceinline__ int lane_id() { return threadIdx.x & 31; }

__device__ __forceinline__ float gelu_f(float x) {
    return 0.5f * x * (1.0f + erff(x * 0.70710678118654752f));
}

// A fragment: 16x32 f16, row-major source with leading dim lda.
// lane l: row = l&15 ; K = ((l>>4)*16) + {0..15}
__device__ __forceinline__ v16h load_a_frag(const _Float16* __restrict__ A, int lda) {
    int l   = lane_id();
    int row = l & 15;
    int kb  = (l >> 4) << 4;                 // 0 or 16
    const _Float16* p = A + (size_t)row * lda + kb;
    v16h a;
#pragma unroll
    for (int i = 0; i < 16; ++i) a[i] = p[i];
    return a;
}

// B fragment: 32x16 where storage is N x K row-major (B^T), contiguous per lane.
// lane l: col n = l&15 ; K = ((l>>4)*16) + {0..15}
__device__ __forceinline__ v16h load_b_frag_nk(const _Float16* __restrict__ Bt, int ldb) {
    int l  = lane_id();
    int n  = l & 15;
    int kb = (l >> 4) << 4;                  // 0 or 16
    const _Float16* p = Bt + (size_t)n * ldb + kb;
    v16h b;
#pragma unroll
    for (int i = 0; i < 16; ++i) b[i] = p[i];
    return b;
}

__device__ __forceinline__ v8f wmma_f16(v16h a, v16h b, v8f c) {
    return __builtin_amdgcn_wmma_f32_16x16x32_f16(false, a, false, b, (short)0, c, false, false);
}

// ---------------------------------------------------------------------------
// LDS-tiled transpose + cast: out[c*ldo + r] = (f16) in[r*ldi + c]
// block (32,8), grid (cols/32, rows/32)
// ---------------------------------------------------------------------------
__global__ void transpose_cast_f32f16(const float* __restrict__ in, int ldi,
                                      _Float16* __restrict__ out, int ldo) {
    __shared__ float tile[32][33];
    int c0 = blockIdx.x * 32, r0 = blockIdx.y * 32;
    int tx = threadIdx.x, ty = threadIdx.y;
#pragma unroll
    for (int j = 0; j < 32; j += 8)
        tile[ty + j][tx] = in[(size_t)(r0 + ty + j) * ldi + (c0 + tx)];
    __syncthreads();
#pragma unroll
    for (int j = 0; j < 32; j += 8)
        out[(size_t)(c0 + ty + j) * ldo + (r0 + tx)] = (_Float16)tile[tx][ty + j];
}

__global__ void transpose_f16(const _Float16* __restrict__ in, int ldi,
                              _Float16* __restrict__ out, int ldo) {
    __shared__ _Float16 tile[32][34];
    int c0 = blockIdx.x * 32, r0 = blockIdx.y * 32;
    int tx = threadIdx.x, ty = threadIdx.y;
#pragma unroll
    for (int j = 0; j < 32; j += 8)
        tile[ty + j][tx] = in[(size_t)(r0 + ty + j) * ldi + (c0 + tx)];
    __syncthreads();
#pragma unroll
    for (int j = 0; j < 32; j += 8)
        out[(size_t)(c0 + ty + j) * ldo + (r0 + tx)] = tile[tx][ty + j];
}

// ---------------------------------------------------------------------------
// LayerNorm over 512 cols, one wave per row, f16 output.
// ---------------------------------------------------------------------------
__global__ void ln512_kernel(const float* __restrict__ x,
                             const float* __restrict__ g,
                             const float* __restrict__ b,
                             _Float16* __restrict__ out, int ldo, int rows) {
    int wave = (int)((blockIdx.x * blockDim.x + threadIdx.x) >> 5);
    int lane = lane_id();
    if (wave >= rows) return;
    const float* xr = x + (size_t)wave * DIM;
    float v[16];
    float s = 0.f;
#pragma unroll
    for (int i = 0; i < 16; ++i) { v[i] = xr[lane * 16 + i]; s += v[i]; }
#pragma unroll
    for (int off = 16; off > 0; off >>= 1) s += __shfl_xor(s, off, 32);
    float mean = s * (1.0f / DIM);
    float vs = 0.f;
#pragma unroll
    for (int i = 0; i < 16; ++i) { float d = v[i] - mean; vs += d * d; }
#pragma unroll
    for (int off = 16; off > 0; off >>= 1) vs += __shfl_xor(vs, off, 32);
    float rstd = rsqrtf(vs * (1.0f / DIM) + 1e-5f);
#pragma unroll
    for (int i = 0; i < 16; ++i) {
        int c = lane * 16 + i;
        out[(size_t)wave * ldo + c] = (_Float16)((v[i] - mean) * rstd * g[c] + b[c]);
    }
}

// ---------------------------------------------------------------------------
// WMMA GEMM: out = act(A(MxK,f16) @ B(KxN,f16 given as B^T NxK) + bias)
// one 16x64 output strip per wave (A frag reused 4x); K-loop step 32,
// software-pipelined (fragments for step k+32 load before step-k WMMAs).
// act: 0 none, 1 gelu, 2 sigmoid. out32/out16 optional.
// grid (N/64, SEQ/64), block 128.
// ---------------------------------------------------------------------------
__global__ void gemm_wmma(const _Float16* __restrict__ A, int lda,
                          const _Float16* __restrict__ Bt, int ldbt,
                          const float* __restrict__ bias,
                          float* __restrict__ out32, int ldo32,
                          _Float16* __restrict__ out16, int ldo16,
                          int K, int act) {
    int widx = threadIdx.x >> 5;
    int lane = lane_id();
    int mt = blockIdx.y * 4 + widx;          // 16-row tile
    int n0 = blockIdx.x * 64;                // 64-col strip
    const _Float16* Ab  = A + (size_t)(mt * 16) * lda;
    const _Float16* Bt0 = Bt + (size_t)n0 * ldbt;
    const _Float16* Bt1 = Bt0 + (size_t)16 * ldbt;
    const _Float16* Bt2 = Bt0 + (size_t)32 * ldbt;
    const _Float16* Bt3 = Bt0 + (size_t)48 * ldbt;

    v8f acc[4] = {};
    // prologue: fragments for k0 = 0
    v16h a  = load_a_frag(Ab, lda);
    v16h b0 = load_b_frag_nk(Bt0, ldbt);
    v16h b1 = load_b_frag_nk(Bt1, ldbt);
    v16h b2 = load_b_frag_nk(Bt2, ldbt);
    v16h b3 = load_b_frag_nk(Bt3, ldbt);

    for (int k0 = 0; k0 < K - 32; k0 += 32) {
        int kn = k0 + 32;
        __builtin_prefetch(Ab + kn + 32, 0, 0);            // global_prefetch_b8
        // issue next-step loads BEFORE consuming current fragments
        v16h an  = load_a_frag(Ab + kn, lda);
        v16h bn0 = load_b_frag_nk(Bt0 + kn, ldbt);
        v16h bn1 = load_b_frag_nk(Bt1 + kn, ldbt);
        v16h bn2 = load_b_frag_nk(Bt2 + kn, ldbt);
        v16h bn3 = load_b_frag_nk(Bt3 + kn, ldbt);
        acc[0] = wmma_f16(a, b0, acc[0]);
        acc[1] = wmma_f16(a, b1, acc[1]);
        acc[2] = wmma_f16(a, b2, acc[2]);
        acc[3] = wmma_f16(a, b3, acc[3]);
        a = an; b0 = bn0; b1 = bn1; b2 = bn2; b3 = bn3;
    }
    acc[0] = wmma_f16(a, b0, acc[0]);
    acc[1] = wmma_f16(a, b1, acc[1]);
    acc[2] = wmma_f16(a, b2, acc[2]);
    acc[3] = wmma_f16(a, b3, acc[3]);

    int rb = (lane >> 4) << 3;
#pragma unroll
    for (int t = 0; t < 4; ++t) {
        int col = n0 + t * 16 + (lane & 15);
        float bv = bias ? bias[col] : 0.f;
#pragma unroll
        for (int r = 0; r < 8; ++r) {
            float vv = acc[t][r] + bv;
            if (act == 1)      vv = gelu_f(vv);
            else if (act == 2) vv = 1.0f / (1.0f + __expf(-vv));
            int row = mt * 16 + rb + r;
            if (out32) out32[(size_t)row * ldo32 + col] = vv;
            if (out16) out16[(size_t)row * ldo16 + col] = (_Float16)vv;
        }
    }
}

// ---------------------------------------------------------------------------
// Sliding-window causal attention, one wave per (head, 16-query tile).
// qkv layout: (SEQ, 3*DIM) f16; q at col h*64, k at 512+h*64.
// vT layout: (DIM, SEQ) f16  (dim-major V for contiguous B fragments).
// Online softmax across 32-key steps; 4 WMMA for QK^T, 4 WMMA for P@V.
// ---------------------------------------------------------------------------
__global__ void attn_swa_wmma(const _Float16* __restrict__ qkv,
                              const _Float16* __restrict__ vT,
                              _Float16* __restrict__ out) {
    __shared__ _Float16 Plds[4][16 * 32];
    const int ldq = 3 * DIM;
    int widx = threadIdx.x >> 5;
    int lane = lane_id();
    int w    = blockIdx.x * 4 + widx;
    int head = w & (HEADS - 1);
    int qt   = w >> 3;
    int q0   = qt * 16;

    const _Float16* qb = qkv + (size_t)q0 * ldq + head * HEAD_DIM;
    v16h qa0 = load_a_frag(qb, ldq);
    v16h qa1 = load_a_frag(qb + 32, ldq);

    float m[8], lsum[8];
    v8f o[4] = {};
#pragma unroll
    for (int r = 0; r < 8; ++r) { m[r] = -1e30f; lsum[r] = 0.f; }

    int col = lane & 15;
    int rb  = (lane >> 4) << 3;
    const float scale = 0.125f;              // 64^-0.5
    _Float16* P = Plds[widx];

    int lo  = q0 - (WINDOW - 1);
    int kt0 = lo > 0 ? (lo & ~31) : 0;

    for (int kt = kt0; kt <= q0 + 15; kt += 32) {
        const _Float16* kb = qkv + (size_t)kt * ldq + DIM + head * HEAD_DIM;
        v16h kb0a = load_b_frag_nk(kb, ldq);                 // keys kt..+15, dims 0..31
        v16h kb0b = load_b_frag_nk(kb + 32, ldq);            // dims 32..63
        v16h kb1a = load_b_frag_nk(kb + (size_t)16 * ldq, ldq);
        v16h kb1b = load_b_frag_nk(kb + (size_t)16 * ldq + 32, ldq);
        v8f s0 = {}, s1 = {};
        s0 = wmma_f16(qa0, kb0a, s0);
        s0 = wmma_f16(qa1, kb0b, s0);
        s1 = wmma_f16(qa0, kb1a, s1);
        s1 = wmma_f16(qa1, kb1b, s1);

#pragma unroll
        for (int r = 0; r < 8; ++r) {
            int q  = q0 + rb + r;
            int d0 = q - (kt + col);
            int d1 = q - (kt + 16 + col);
            bool ok0 = (d0 >= 0) && (d0 < WINDOW);
            bool ok1 = (d1 >= 0) && (d1 < WINDOW);
            float v0 = ok0 ? s0[r] * scale : -3e38f;
            float v1 = ok1 ? s1[r] * scale : -3e38f;
            float rm = fmaxf(v0, v1);
#pragma unroll
            for (int off = 1; off < 16; off <<= 1) rm = fmaxf(rm, __shfl_xor(rm, off, 32));
            float mnew  = fmaxf(m[r], rm);
            float alpha = __expf(m[r] - mnew);
            float p0 = ok0 ? __expf(v0 - mnew) : 0.f;
            float p1 = ok1 ? __expf(v1 - mnew) : 0.f;
            float rs = p0 + p1;
#pragma unroll
            for (int off = 1; off < 16; off <<= 1) rs += __shfl_xor(rs, off, 32);
            lsum[r] = lsum[r] * alpha + rs;
            m[r]    = mnew;
#pragma unroll
            for (int t = 0; t < 4; ++t) o[t][r] *= alpha;
            P[(rb + r) * 32 + col]      = (_Float16)p0;
            P[(rb + r) * 32 + col + 16] = (_Float16)p1;
        }
        // wave-internal LDS RAW: stores above feed cross-lane loads below
        asm volatile("s_wait_dscnt 0" ::: "memory");
        v16h pa = load_a_frag(P, 32);
#pragma unroll
        for (int t = 0; t < 4; ++t) {
            // B(k=key, n=dim) from dim-major vT: contiguous 16 keys per lane
            const _Float16* vb = vT + (size_t)(head * HEAD_DIM + t * 16) * SEQ + kt;
            v16h bf = load_b_frag_nk(vb, SEQ);
            o[t] = wmma_f16(pa, bf, o[t]);
        }
    }

#pragma unroll
    for (int r = 0; r < 8; ++r) {
        float inv = 1.0f / lsum[r];
        int row = q0 + rb + r;
#pragma unroll
        for (int t = 0; t < 4; ++t)
            out[(size_t)row * DIM + head * HEAD_DIM + t * 16 + col] = (_Float16)(o[t][r] * inv);
    }
}

// ---------------------------------------------------------------------------
// Elementwise kernels
// ---------------------------------------------------------------------------
__global__ void gate_combine(const float* __restrict__ x, const float* __restrict__ s,
                             const float* __restrict__ l, const float* __restrict__ g,
                             float* __restrict__ xres, int n) {
    int i = blockIdx.x * blockDim.x + threadIdx.x;
    if (i < n) {
        float gv = g[i];
        xres[i] = x[i] + (1.0f - gv) * s[i] + gv * l[i];
    }
}

__global__ void add_out(const float* __restrict__ a, const float* __restrict__ b,
                        float* __restrict__ out, int n) {
    int i = blockIdx.x * blockDim.x + threadIdx.x;
    if (i < n) out[i] = a[i] + b[i];
}

// ---------------------------------------------------------------------------
// Host orchestration
// ---------------------------------------------------------------------------
extern "C" void kernel_launch(void* const* d_in, const int* in_sizes, int n_in,
                              void* d_out, int out_size, void* d_ws, size_t ws_size,
                              hipStream_t stream) {
    (void)in_sizes; (void)n_in; (void)out_size; (void)ws_size;

    const float* x     = (const float*)d_in[0];
    const float* ln1_g = (const float*)d_in[1];
    const float* ln1_b = (const float*)d_in[2];
    const float* ln2_g = (const float*)d_in[3];
    const float* ln2_b = (const float*)d_in[4];
    const float* wqkv  = (const float*)d_in[5];
    const float* bqkv  = (const float*)d_in[6];
    const float* wproj = (const float*)d_in[7];
    const float* bproj = (const float*)d_in[8];
    const float* wm1   = (const float*)d_in[9];
    const float* bm1   = (const float*)d_in[10];
    const float* wm2   = (const float*)d_in[11];
    const float* bm2   = (const float*)d_in[12];
    const float* wg1   = (const float*)d_in[13];
    const float* bg1   = (const float*)d_in[14];
    const float* wg2   = (const float*)d_in[15];
    const float* bg2   = (const float*)d_in[16];
    const float* wf1   = (const float*)d_in[17];
    const float* bf1   = (const float*)d_in[18];
    const float* wf2   = (const float*)d_in[19];
    const float* bf2   = (const float*)d_in[20];
    float* outp = (float*)d_out;

    // --- workspace carve ---
    char* base = (char*)d_ws;
    size_t off = 0;
    auto carve = [&](size_t bytes) {
        void* p = base + off;
        off += (bytes + 255) & ~(size_t)255;
        return p;
    };
    _Float16* qkv16  = (_Float16*)carve((size_t)SEQ * 3 * DIM * 2);
    _Float16* cat16  = (_Float16*)carve((size_t)SEQ * 3 * DIM * 2);   // [short | long | xn]
    _Float16* vT16   = (_Float16*)carve((size_t)DIM * SEQ * 2);       // dim-major V
    _Float16* attn16 = (_Float16*)carve((size_t)SEQ * DIM * 2);
    _Float16* h16    = (_Float16*)carve((size_t)SEQ * MEM_H * 2);
    _Float16* g116   = (_Float16*)carve((size_t)SEQ * DIM * 2);
    _Float16* y16    = (_Float16*)carve((size_t)SEQ * DIM * 2);
    _Float16* f116   = (_Float16*)carve((size_t)SEQ * FFN_H * 2);
    float* short32   = (float*)carve((size_t)SEQ * DIM * 4);
    float* long32    = (float*)carve((size_t)SEQ * DIM * 4);
    float* gate32    = (float*)carve((size_t)SEQ * DIM * 4);
    float* xres32    = (float*)carve((size_t)SEQ * DIM * 4);
    float* f232      = (float*)carve((size_t)SEQ * DIM * 4);
    // transposed f16 weights (N x K each)
    _Float16* wqkvT  = (_Float16*)carve((size_t)3 * DIM * DIM * 2);
    _Float16* wprojT = (_Float16*)carve((size_t)DIM * DIM * 2);
    _Float16* wm1T   = (_Float16*)carve((size_t)MEM_H * DIM * 2);
    _Float16* wm2T   = (_Float16*)carve((size_t)DIM * MEM_H * 2);
    _Float16* wg1T   = (_Float16*)carve((size_t)DIM * 3 * DIM * 2);
    _Float16* wg2T   = (_Float16*)carve((size_t)DIM * DIM * 2);
    _Float16* wf1T   = (_Float16*)carve((size_t)FFN_H * DIM * 2);
    _Float16* wf2T   = (_Float16*)carve((size_t)DIM * FFN_H * 2);

    // weight transpose-casts: in (K x N) f32 -> out (N x K) f16
    auto tcast = [&](const float* src, _Float16* dst, int K, int N) {
        dim3 grid(N / 32, K / 32), block(32, 8);
        transpose_cast_f32f16<<<grid, block, 0, stream>>>(src, N, dst, K);
    };
    tcast(wqkv, wqkvT, DIM, 3 * DIM);
    tcast(wproj, wprojT, DIM, DIM);
    tcast(wm1, wm1T, DIM, MEM_H);
    tcast(wm2, wm2T, MEM_H, DIM);
    tcast(wg1, wg1T, 3 * DIM, DIM);
    tcast(wg2, wg2T, DIM, DIM);
    tcast(wf1, wf1T, DIM, FFN_H);
    tcast(wf2, wf2T, FFN_H, DIM);

    _Float16* xn16 = cat16 + 2 * DIM;   // xn lives in cat[:, 1024:1536], ld = 1536
    const int ldcat = 3 * DIM;

    // LN1 -> xn (f16, inside concat buffer)
    ln512_kernel<<<SEQ / 8, 256, 0, stream>>>(x, ln1_g, ln1_b, xn16, ldcat, SEQ);

    auto gemm = [&](const _Float16* A, int lda, const _Float16* Bt,
                    const float* bias, float* o32, int ldo32,
                    _Float16* o16, int ldo16, int N, int K, int act) {
        dim3 grid(N / 64, SEQ / 64);
        gemm_wmma<<<grid, 128, 0, stream>>>(A, lda, Bt, K, bias, o32, ldo32, o16, ldo16, K, act);
    };

    // qkv = xn @ wqkv + bqkv  (f16)
    gemm(xn16, ldcat, wqkvT, bqkv, nullptr, 0, qkv16, 3 * DIM, 3 * DIM, DIM, 0);

    // dim-major V for attention B fragments
    {
        dim3 grid(DIM / 32, SEQ / 32), block(32, 8);
        transpose_f16<<<grid, block, 0, stream>>>(qkv16 + 2 * DIM, 3 * DIM, vT16, SEQ);
    }

    // sliding-window attention -> attn16
    attn_swa_wmma<<<(SEQ / 16) * HEADS / 4, 128, 0, stream>>>(qkv16, vT16, attn16);

    // short = attn @ wproj + bproj  -> f32 + f16 into cat[:, 0:512]
    gemm(attn16, DIM, wprojT, bproj, short32, DIM, cat16, ldcat, DIM, DIM, 0);

    // long-term memory MLP: h = gelu(xn @ wm1 + bm1); long = h @ wm2 + bm2
    gemm(xn16, ldcat, wm1T, bm1, nullptr, 0, h16, MEM_H, MEM_H, DIM, 1);
    gemm(h16, MEM_H, wm2T, bm2, long32, DIM, cat16 + DIM, ldcat, DIM, MEM_H, 0);

    // gate = sigmoid(gelu(cat @ wg1 + bg1) @ wg2 + bg2)
    gemm(cat16, ldcat, wg1T, bg1, nullptr, 0, g116, DIM, DIM, 3 * DIM, 1);
    gemm(g116, DIM, wg2T, bg2, gate32, DIM, nullptr, 0, DIM, DIM, 2);

    // xres = x + (1-g)*short + g*long
    int ntot = SEQ * DIM;
    gate_combine<<<(ntot + 255) / 256, 256, 0, stream>>>(x, short32, long32, gate32, xres32, ntot);

    // FFN: out = xres + gelu(ln2(xres) @ wf1 + bf1) @ wf2 + bf2
    ln512_kernel<<<SEQ / 8, 256, 0, stream>>>(xres32, ln2_g, ln2_b, y16, DIM, SEQ);
    gemm(y16, DIM, wf1T, bf1, nullptr, 0, f116, FFN_H, FFN_H, DIM, 1);
    gemm(f116, FFN_H, wf2T, bf2, f232, DIM, nullptr, 0, DIM, FFN_H, 0);
    add_out<<<(ntot + 255) / 256, 256, 0, stream>>>(xres32, f232, outp, ntot);
}
